// RNNCore_52673478918341
// MI455X (gfx1250) — matrix operand order repaired and checked by reading, not compiled, not run on hardware
//
#include <hip/hip_runtime.h>

// B=131072, IN=128, H=256 fused  relu(x@Wp^T+bp) -> GRU cell, bf16 WMMA path.

#define BATCH 131072
#define INF   128
#define HID   256

typedef __attribute__((ext_vector_type(8)))  __bf16 v8bf;
typedef __attribute__((ext_vector_type(16))) __bf16 v16bf;
typedef __attribute__((ext_vector_type(4)))  __bf16 v4bf;
typedef __attribute__((ext_vector_type(8)))  float  v8f;
typedef __attribute__((ext_vector_type(16))) float  v16f;
typedef __attribute__((ext_vector_type(4)))  float  v4f;
typedef int vi4 __attribute__((vector_size(16)));   // matches builtin's pointee type

#if __has_builtin(__builtin_amdgcn_global_load_async_to_lds_b128) && __has_builtin(__builtin_amdgcn_s_wait_asynccnt)
#define HAVE_ASYNC_LDS 1
#else
#define HAVE_ASYNC_LDS 0
#endif

#if HAVE_ASYNC_LDS
__device__ __forceinline__ void async_cp16(const void* g, void* l) {
    __builtin_amdgcn_global_load_async_to_lds_b128(
        (__attribute__((address_space(1))) vi4*)(g),
        (__attribute__((address_space(3))) vi4*)(l), 0, 0);
}
#endif

__device__ __forceinline__ v16bf cat16(v8bf lo, v8bf hi) {
    return __builtin_shufflevector(lo, hi, 0,1,2,3,4,5,6,7,8,9,10,11,12,13,14,15);
}

// A fragment (16x32 bf16) from f32 row-major; per-lane: K chunks {k0+8hs..} and {k0+16+8hs..}
__device__ __forceinline__ v16bf load_a_f32(const float* __restrict__ row, int k0, int hs) {
    const float* c0 = row + k0 + 8 * hs;
    const float* c1 = row + k0 + 16 + 8 * hs;
    float4 f0 = *(const float4*)(c0);
    float4 f1 = *(const float4*)(c0 + 4);
    float4 f2 = *(const float4*)(c1);
    float4 f3 = *(const float4*)(c1 + 4);
    v16f vf = { f0.x, f0.y, f0.z, f0.w, f1.x, f1.y, f1.z, f1.w,
                f2.x, f2.y, f2.z, f2.w, f3.x, f3.y, f3.z, f3.w };
    return __builtin_convertvector(vf, v16bf);   // packed v_cvt_pk_bf16_f32 path
}

// A fragment from bf16 row-major (LDS)
__device__ __forceinline__ v16bf load_a_bf(const __bf16* row, int k0, int hs) {
    v8bf lo = *(const v8bf*)(row + k0 + 8 * hs);
    v8bf hi = *(const v8bf*)(row + k0 + 16 + 8 * hs);
    return cat16(lo, hi);
}

// B fragment (32x16 bf16): per-lane column = weight row; K chunks {k0+8hs..} and {k0+16+8hs..}
__device__ __forceinline__ v16bf load_b_bf(const __bf16* colrow, int k0, int hs) {
    const __bf16* p = colrow + k0 + 8 * hs;
    v8bf lo = *(const v8bf*)(p);
    v8bf hi = *(const v8bf*)(p + 16);
    return cat16(lo, hi);
}

__device__ __forceinline__ float sigmoidf(float x) { return 1.0f / (1.0f + __expf(-x)); }
__device__ __forceinline__ float tanhf_fast(float x) {
    float ax = fabsf(x);
    float e = __expf(-2.0f * ax);
    float t = (1.0f - e) / (1.0f + e);
    return copysignf(t, x);
}

// ---- weight fp32 -> bf16 conversion into d_ws ----
// layout (bf16 elems): Wp @0 (32768), W_ih @32768 (196608), W_hh @229376 (196608)
__global__ __launch_bounds__(256) void cvt_weights(const float* __restrict__ wp,
                                                   const float* __restrict__ wih,
                                                   const float* __restrict__ whh,
                                                   __bf16* __restrict__ ws) {
    const int N0 = 32768, N1 = 196608;
    int base = (blockIdx.x * 256 + threadIdx.x) * 4;
    const float* src;
    if (base < N0)            src = wp  + base;
    else if (base < N0 + N1)  src = wih + (base - N0);
    else if (base < N0+2*N1)  src = whh + (base - N0 - N1);
    else return;
    float4 f = *(const float4*)src;
    v4f vf = { f.x, f.y, f.z, f.w };
    *(v4bf*)(ws + base) = __builtin_convertvector(vf, v4bf);
}

// ---- fused kernel: 8 waves, 128 rows per workgroup ----
__global__ __launch_bounds__(256) void gru_fused(
    const float* __restrict__ x, const float* __restrict__ h,
    const float* __restrict__ bp, const float* __restrict__ b_ih, const float* __restrict__ b_hh,
    const __bf16* __restrict__ wp_bf, const __bf16* __restrict__ wih_bf, const __bf16* __restrict__ whh_bf,
    float* __restrict__ out)
{
    // padded strides: 136 elems (272B -> rows offset by 4 banks), 264 elems (528B -> 4 banks)
    __shared__ __bf16 p_lds[128 * 264];   // 67584 B : p tile, rows local 0..127
    __shared__ __bf16 w_lds[256 * 136];   // 69632 B : Wp staging, then [96][264] slices

    const int tid  = threadIdx.x;
    const int wv   = tid >> 5;
    const int lane = tid & 31;
    const int ln   = lane & 15;
    const int hs   = lane >> 4;
    const int rowBase = blockIdx.x * 128;
    const int myRow = rowBase + wv * 16 + ln;

    // stage Wp (bf16) into LDS with padded stride 136 (row = 16x16B src, 17x16B dst)
    {
        uint4* dst = (uint4*)w_lds;
        const uint4* src = (const uint4*)wp_bf;
        for (int i = tid; i < 4096; i += 256) {
            int r = i >> 4, q = i & 15;
#if HAVE_ASYNC_LDS
            async_cp16(src + i, dst + r * 17 + q);
#else
            dst[r * 17 + q] = src[i];
#endif
        }
#if HAVE_ASYNC_LDS
        __builtin_amdgcn_s_wait_asynccnt(0);
#endif
    }
    __syncthreads();

    // ---- stage 1: p = relu(x @ Wp^T + bp) ----
    const float* xrow = x + (size_t)myRow * INF;
    v16bf xa[4];
#pragma unroll
    for (int kk = 0; kk < 4; ++kk) xa[kk] = load_a_f32(xrow, kk * 32, hs);

#pragma unroll 1
    for (int ct = 0; ct < 16; ct += 2) {   // two independent accum chains hide WMMA RAW hazard
        v8f acc0 = {}, acc1 = {};
        const __bf16* wb0 = w_lds + (ct * 16 + ln) * 136;
        const __bf16* wb1 = w_lds + (ct * 16 + 16 + ln) * 136;
#pragma unroll
        for (int kk = 0; kk < 4; ++kk) {
            v16bf b0 = load_b_bf(wb0, kk * 32, hs);
            acc0 = __builtin_amdgcn_wmma_f32_16x16x32_bf16(false, xa[kk], false, b0,
                                                           (short)0, acc0, false, false);
            v16bf b1 = load_b_bf(wb1, kk * 32, hs);
            acc1 = __builtin_amdgcn_wmma_f32_16x16x32_bf16(false, xa[kk], false, b1,
                                                           (short)0, acc1, false, false);
        }
        float bv0 = bp[ct * 16 + ln];
        float bv1 = bp[ct * 16 + 16 + ln];
        __bf16* pdst0 = p_lds + (wv * 16) * 264 + ct * 16 + ln;
#pragma unroll
        for (int i = 0; i < 8; ++i) {
            float va = fmaxf(acc0[i] + bv0, 0.f);
            float vb = fmaxf(acc1[i] + bv1, 0.f);
            pdst0[(i + 8 * hs) * 264]      = (__bf16)va;
            pdst0[(i + 8 * hs) * 264 + 16] = (__bf16)vb;
        }
    }

    // ---- h A-fragments held in registers (reused across 16 column groups) ----
    const float* hrow = h + (size_t)myRow * HID;
    v16bf ha[8];
#pragma unroll
    for (int kk = 0; kk < 8; ++kk) ha[kk] = load_a_f32(hrow, kk * 32, hs);

    const size_t BH = (size_t)BATCH * HID;

#pragma unroll 1
    for (int c = 0; c < 16; ++c) {
        __syncthreads();  // previous w_lds reads done
        // stage 6x16x256 bf16 weight slices: m=0..2 -> W_ih gates r,z,n ; m=3..5 -> W_hh
        {
            uint4* dst = (uint4*)w_lds;
            for (int i = tid; i < 3072; i += 256) {
                int r = i >> 5, q = i & 31;
                int m = r >> 4, n = r & 15;
                const __bf16* wsrc = (m < 3 ? wih_bf : whh_bf);
                const uint4* src = (const uint4*)(wsrc + (size_t)((m % 3) * 256 + c * 16 + n) * 256);
#if HAVE_ASYNC_LDS
                async_cp16(src + q, dst + r * 33 + q);
#else
                dst[r * 33 + q] = src[q];
#endif
            }
#if HAVE_ASYNC_LDS
            __builtin_amdgcn_s_wait_asynccnt(0);
#endif
        }
        __syncthreads();

        v8f ar = {}, az = {}, an = {}, br = {}, bz = {}, bn = {};
        const __bf16* prow = p_lds + (wv * 16 + ln) * 264;
#pragma unroll
        for (int kk = 0; kk < 8; ++kk) {
            v16bf ap = load_a_bf(prow, kk * 32, hs);
            v16bf ah = ha[kk];
            v16bf w0 = load_b_bf(w_lds + (0 * 16 + ln) * 264, kk * 32, hs);
            ar = __builtin_amdgcn_wmma_f32_16x16x32_bf16(false, ap, false, w0, (short)0, ar, false, false);
            v16bf w1 = load_b_bf(w_lds + (1 * 16 + ln) * 264, kk * 32, hs);
            az = __builtin_amdgcn_wmma_f32_16x16x32_bf16(false, ap, false, w1, (short)0, az, false, false);
            v16bf w2 = load_b_bf(w_lds + (2 * 16 + ln) * 264, kk * 32, hs);
            an = __builtin_amdgcn_wmma_f32_16x16x32_bf16(false, ap, false, w2, (short)0, an, false, false);
            v16bf w3 = load_b_bf(w_lds + (3 * 16 + ln) * 264, kk * 32, hs);
            br = __builtin_amdgcn_wmma_f32_16x16x32_bf16(false, ah, false, w3, (short)0, br, false, false);
            v16bf w4 = load_b_bf(w_lds + (4 * 16 + ln) * 264, kk * 32, hs);
            bz = __builtin_amdgcn_wmma_f32_16x16x32_bf16(false, ah, false, w4, (short)0, bz, false, false);
            v16bf w5 = load_b_bf(w_lds + (5 * 16 + ln) * 264, kk * 32, hs);
            bn = __builtin_amdgcn_wmma_f32_16x16x32_bf16(false, ah, false, w5, (short)0, bn, false, false);
        }

        // gates + blend; D layout: elem i -> row = base+i+8*hs, col = c*16+ln
        int col = c * 16 + ln;
        float bir = b_ih[col], biz = b_ih[256 + col], bin = b_ih[512 + col];
        float bhr = b_hh[col], bhz = b_hh[256 + col], bhn = b_hh[512 + col];
#pragma unroll
        for (int i = 0; i < 8; ++i) {
            int row = rowBase + wv * 16 + i + 8 * hs;
            float rg = sigmoidf((ar[i] + bir) + (br[i] + bhr));
            float zg = sigmoidf((az[i] + biz) + (bz[i] + bhz));
            float nn = tanhf_fast((an[i] + bin) + rg * (bn[i] + bhn));
            size_t off = (size_t)row * HID + col;
            float h0 = h[off];
            float o = (1.f - zg) * nn + zg * h0;
            __builtin_nontemporal_store(o, out + off);
            __builtin_nontemporal_store(o, out + BH + off);
        }
    }
}

extern "C" void kernel_launch(void* const* d_in, const int* in_sizes, int n_in,
                              void* d_out, int out_size, void* d_ws, size_t ws_size,
                              hipStream_t stream) {
    const float* x   = (const float*)d_in[0];
    const float* h   = (const float*)d_in[1];
    const float* Wp  = (const float*)d_in[2];
    const float* bp  = (const float*)d_in[3];
    const float* Wih = (const float*)d_in[4];
    const float* Whh = (const float*)d_in[5];
    const float* bih = (const float*)d_in[6];
    const float* bhh = (const float*)d_in[7];
    float* out = (float*)d_out;
    __bf16* wsb = (__bf16*)d_ws;   // needs 851968 bytes

    // 425984 bf16 weight elems / 4 per thread / 256 per block = 416 blocks
    cvt_weights<<<416, 256, 0, stream>>>(Wp, Wih, Whh, wsb);
    gru_fused<<<BATCH / 128, 256, 0, stream>>>(x, h, bp, bih, bhh,
                                               wsb, wsb + 32768, wsb + 229376, out);
}
